// ContrastiveMambaEncoder_446676599145
// MI455X (gfx1250) — compile-verified, matrix operand order checked by reading, not run on hardware
//
#include <hip/hip_runtime.h>

// ---------------------------------------------------------------------------
// ContrastiveMambaEncoder for MI455X (gfx1250, wave32, WMMA + TDM)
//
// bf16 operands + f32 accumulate via V_WMMA_F32_16X16X32_BF16 for the three
// large GEMMs (~60 of 62 GFLOP). Large-N GEMMs stage the shared B tile in LDS
// with the Tensor Data Mover (double-buffered, TENSORcnt-synchronized), so the
// 8 waves of a block read B from LDS instead of 8x-redundant global loads.
// ---------------------------------------------------------------------------

typedef __bf16 bf16x16 __attribute__((ext_vector_type(16)));
typedef float  f32x8   __attribute__((ext_vector_type(8)));
typedef unsigned int u32x4 __attribute__((ext_vector_type(4)));
typedef int          i32x4 __attribute__((ext_vector_type(4)));
typedef int          i32x8 __attribute__((ext_vector_type(8)));
typedef unsigned short u16;
typedef unsigned int   u32;

#define D_MODEL   768
#define D_STATE   16
#define D_CONV    4
#define D_INNER   1536
#define DT_RANK   48
#define EMBED_DIM 256
#define BATCH     4
#define SEQ       2048
#define ROWS      (BATCH * SEQ)            // 8192
#define XDBL_W    (DT_RANK + 2 * D_STATE)  // 80

__device__ __forceinline__ u16 f2bf(float f) {
  u32 x = __float_as_uint(f);
  x += 0x7FFFu + ((x >> 16) & 1u);     // round-to-nearest-even
  return (u16)(x >> 16);
}

__device__ __forceinline__ f32x8 wmma_bf16(const bf16x16& a, const bf16x16& b,
                                           const f32x8& c) {
  return __builtin_amdgcn_wmma_f32_16x16x32_bf16(false, a, false, b, (short)0,
                                                 c, false, false);
}

// --------------------------- elementwise kernels ---------------------------

__global__ void k_f32_to_bf16(const float* __restrict__ in,
                              u16* __restrict__ out, int n) {
  int i = blockIdx.x * blockDim.x + threadIdx.x;
  if (i < n) out[i] = f2bf(in[i]);
}

__global__ void k_gather_embed(const int* __restrict__ ids,
                               const float* __restrict__ emb,
                               u16* __restrict__ x, int n) {
  int i = blockIdx.x * blockDim.x + threadIdx.x;
  if (i >= n) return;
  int row = i / D_MODEL, c = i - row * D_MODEL;
  x[i] = f2bf(emb[(size_t)ids[row] * D_MODEL + c]);
}

// ---------------------- TDM: async 64x64 bf16 tile -> LDS ------------------
// D# per CDNA5 ISA: 2D tile, data_size=2B,
// tile_dim0 = 64 (contiguous K), tile_dim1 = 64 (rows), row stride = K elems.
// Groups 2/3 zeroed (<=2D tensor). Tracked with TENSORcnt.

__device__ __forceinline__ void tdm_load_b_tile(const u16* gaddr,
                                                unsigned lds_off, int K) {
  unsigned long long ga = (unsigned long long)(size_t)gaddr;
  unsigned td0 = (unsigned)K;            // tensor_dim0 (elements)
  unsigned td1 = 64u;                    // tensor_dim1 (rows)
  unsigned long long s0 = (unsigned long long)(unsigned)K;  // dim0 stride

  u32x4 g0;
  g0[0] = 1u;                                            // count=1 (valid D#)
  g0[1] = lds_off;                                       // LDS byte address
  g0[2] = (u32)(ga & 0xFFFFFFFFull);                     // global_addr lo
  g0[3] = (u32)((ga >> 32) & 0x01FFFFFFull) | (2u << 30);// addr hi | type=2

  i32x8 g1;
  g1[0] = (int)(1u << 16);                               // wg_mask=0, 2B elems
  g1[1] = (int)((td0 & 0xFFFFu) << 16);                  // tensor_dim0[15:0]
  g1[2] = (int)((td0 >> 16) | ((td1 & 0xFFFFu) << 16));  // td0 hi | td1 lo
  g1[3] = (int)((td1 >> 16) | (64u << 16));              // td1 hi | tile_dim0
  g1[4] = (int)64u;                                      // tile_dim1=64, td2=0
  g1[5] = (int)(u32)(s0 & 0xFFFFFFFFull);                // dim0 stride lo
  g1[6] = (int)(u32)((s0 >> 32) & 0xFFFFull);            // stride hi | d1s lo
  g1[7] = 0;                                             // dim1 stride (2D)

  i32x4 z4 = {0, 0, 0, 0};
#if __clang_major__ >= 23
  i32x8 z8 = {0, 0, 0, 0, 0, 0, 0, 0};
  __builtin_amdgcn_tensor_load_to_lds(g0, g1, z4, z4, z8, 0);
#else
  __builtin_amdgcn_tensor_load_to_lds(g0, g1, z4, z4, 0);
#endif
}

// --------------------- WMMA GEMM with TDM-staged B tile --------------------
// C[M,N] = A[M,K] * B[N,K]^T  (bf16 in, f32 out). Requires M%256==0, N%64==0,
// K%64==0. Block: 256 thr = 8 waves stacked in M; wave tile 32(M) x 64(N).
// Wave 0 double-buffers the block's 64-row B tile through the TDM; all waves
// read B fragments from LDS (ds_load_b128), A fragments direct from global.

__global__ __launch_bounds__(256) void k_gemm_bf16_wmma_tdm(
    const u16* __restrict__ A, const u16* __restrict__ B,
    float* __restrict__ C, int M, int N, int K) {
  __shared__ u16 bt[2][64 * 64];

  const int wave = threadIdx.x >> 5;
  const int lane = threadIdx.x & 31;
  const int half = lane >> 4;
  const int lmod = lane & 15;
  const int m0   = blockIdx.x * 256 + wave * 32;
  const int n0   = blockIdx.y * 64;

  const u16* A0 = A + (size_t)(m0 + lmod) * K;
  const u16* A1 = A + (size_t)(m0 + 16 + lmod) * K;
  const u16* Bt = B + (size_t)n0 * K;        // tile origin: row n0, col 0

  union Frag { uint4 u[2]; bf16x16 v; };
  f32x8 acc[2][4] = {};

  const int nkb = K >> 6;                    // 64-wide K blocks
  if (wave == 0)
    tdm_load_b_tile(Bt, (unsigned)(size_t)&bt[0][0], K);

  for (int kb = 0; kb < nkb; ++kb) {
    if (wave == 0) {
      if (kb + 1 < nkb) {
        tdm_load_b_tile(Bt + (size_t)(kb + 1) * 64,
                        (unsigned)(size_t)&bt[(kb + 1) & 1][0], K);
        __builtin_amdgcn_s_wait_tensorcnt(1);   // buffer kb has landed
      } else {
        __builtin_amdgcn_s_wait_tensorcnt(0);   // last buffer has landed
      }
    }
    __syncthreads();                         // B tile kb visible to all waves

    const u16* lb = &bt[kb & 1][0];
    const int  kg = kb * 64;
    __builtin_prefetch(A0 + kg + 128, 0, 3); // next A K-blocks
    __builtin_prefetch(A1 + kg + 128, 0, 3);
#pragma unroll
    for (int ks = 0; ks < 64; ks += 32) {
      Frag fa0, fa1;
      fa0.u[0] = *(const uint4*)(A0 + kg + ks + 8 * half);
      fa0.u[1] = *(const uint4*)(A0 + kg + ks + 16 + 8 * half);
      fa1.u[0] = *(const uint4*)(A1 + kg + ks + 8 * half);
      fa1.u[1] = *(const uint4*)(A1 + kg + ks + 16 + 8 * half);
#pragma unroll
      for (int nt = 0; nt < 4; ++nt) {
        Frag fb;
        const u16* br = lb + (nt * 16 + lmod) * 64 + ks;
        fb.u[0] = *(const uint4*)(br + 8 * half);
        fb.u[1] = *(const uint4*)(br + 16 + 8 * half);
        acc[0][nt] = wmma_bf16(fa0.v, fb.v, acc[0][nt]);
        acc[1][nt] = wmma_bf16(fa1.v, fb.v, acc[1][nt]);
      }
    }
    __syncthreads();                         // done reading buffer kb
  }

  // C/D layout: VGPR i, lane l -> row += i + 8*(l>>4), col = n0 + (l&15)
#pragma unroll
  for (int mt = 0; mt < 2; ++mt) {
#pragma unroll
    for (int nt = 0; nt < 4; ++nt) {
      int col = n0 + nt * 16 + lmod;
      if (col < N) {
#pragma unroll
        for (int i = 0; i < 8; ++i) {
          int row = m0 + mt * 16 + i + 8 * half;
          C[(size_t)row * N + col] = acc[mt][nt][i];
        }
      }
    }
  }
}

// ----------------- WMMA GEMM, direct-global B (ragged N) -------------------
// Used for x_proj (N = 80). M%128==0, K%32==0; N arbitrary.

__global__ __launch_bounds__(256) void k_gemm_bf16_wmma(
    const u16* __restrict__ A, const u16* __restrict__ B,
    float* __restrict__ C, int M, int N, int K) {
  const int wave = threadIdx.x >> 5;
  const int lane = threadIdx.x & 31;
  const int half = lane >> 4;
  const int lmod = lane & 15;
  const int m0   = blockIdx.x * 128 + wave * 16;
  const int n0   = blockIdx.y * 64;

  const u16* Arow = A + (size_t)(m0 + lmod) * K;
  const u16* Brow[4];
#pragma unroll
  for (int nt = 0; nt < 4; ++nt) {
    int nn = n0 + nt * 16 + lmod;
    if (nn >= N) nn = N - 1;                 // clamp; discarded at store
    Brow[nt] = B + (size_t)nn * K;
  }

  union Frag { uint4 u[2]; bf16x16 v; };
  f32x8 acc[4] = {};

  for (int k0 = 0; k0 < K; k0 += 32) {
    Frag fa;
    fa.u[0] = *(const uint4*)(Arow + k0 + 8 * half);
    fa.u[1] = *(const uint4*)(Arow + k0 + 16 + 8 * half);
    __builtin_prefetch(Arow + k0 + 64, 0, 3);
#pragma unroll
    for (int nt = 0; nt < 4; ++nt) {
      Frag fb;
      fb.u[0] = *(const uint4*)(Brow[nt] + k0 + 8 * half);
      fb.u[1] = *(const uint4*)(Brow[nt] + k0 + 16 + 8 * half);
      acc[nt] = wmma_bf16(fa.v, fb.v, acc[nt]);
    }
  }

#pragma unroll
  for (int nt = 0; nt < 4; ++nt) {
    int col = n0 + nt * 16 + lmod;
    if (col < N) {
#pragma unroll
      for (int i = 0; i < 8; ++i) {
        int row = m0 + i + 8 * half;
        C[(size_t)row * N + col] = acc[nt][i];
      }
    }
  }
}

// ----------------------- conv1d (depthwise, causal) + SiLU -----------------

__global__ void k_conv_silu(const float* __restrict__ xz,
                            const float* __restrict__ cw,
                            const float* __restrict__ cb,
                            float* __restrict__ xc,
                            u16* __restrict__ xcb, int n) {
  int i = blockIdx.x * blockDim.x + threadIdx.x;
  if (i >= n) return;
  int row = i / D_INNER, d = i - row * D_INNER;
  int t = row & (SEQ - 1);
  float acc = cb[d];
#pragma unroll
  for (int j = 0; j < D_CONV; ++j) {
    int tt = t - (D_CONV - 1) + j;
    if (tt >= 0)
      acc += cw[d * D_CONV + j] *
             xz[(size_t)(row - (D_CONV - 1) + j) * (2 * D_INNER) + d];
  }
  float s = acc / (1.f + __expf(-acc));     // silu
  xc[i]  = s;
  xcb[i] = f2bf(s);
}

// ------------------------- delta = softplus(dt @ Wdt^T + b) ----------------

__global__ void k_delta(const float* __restrict__ xdbl,
                        const float* __restrict__ dtw,
                        const float* __restrict__ dtb,
                        float* __restrict__ delta, int n) {
  int i = blockIdx.x * blockDim.x + threadIdx.x;
  if (i >= n) return;
  int row = i / D_INNER, d = i - row * D_INNER;
  const float* xr = xdbl + (size_t)row * XDBL_W;
  const float* wr = dtw + (size_t)d * DT_RANK;
  float acc = dtb[d];
#pragma unroll 8
  for (int r = 0; r < DT_RANK; ++r) acc += xr[r] * wr[r];
  delta[i] = (acc > 20.f) ? acc : log1pf(__expf(acc));
}

// ----------------------------- selective scan ------------------------------
// One thread per (b, d) channel: 16 fp32 states in registers, L=2048 steps.
// delta/xc coalesced across lanes; B/C wave-uniform (L0 broadcast).

__global__ __launch_bounds__(256) void k_scan(
    const float* __restrict__ delta, const float* __restrict__ xdbl,
    const float* __restrict__ xc, const float* __restrict__ A_log,
    float* __restrict__ ys) {
  int tid = blockIdx.x * blockDim.x + threadIdx.x;
  if (tid >= BATCH * D_INNER) return;
  int b = tid / D_INNER, d = tid - b * D_INNER;

  float Ad[D_STATE];
#pragma unroll
  for (int nS = 0; nS < D_STATE; ++nS)
    Ad[nS] = -__expf(A_log[(size_t)d * D_STATE + nS]);

  float h[D_STATE];
#pragma unroll
  for (int nS = 0; nS < D_STATE; ++nS) h[nS] = 0.f;

  for (int t = 0; t < SEQ; ++t) {
    size_t row = (size_t)b * SEQ + t;
    float dt = delta[row * D_INNER + d];
    float xv = xc[row * D_INNER + d];
    const float* bc = xdbl + row * XDBL_W;
    float du = dt * xv;
    float y = 0.f;
#pragma unroll
    for (int nS = 0; nS < D_STATE; ++nS) {
      float dA = __expf(dt * Ad[nS]);
      h[nS] = h[nS] * dA + du * bc[DT_RANK + nS];
      y += h[nS] * bc[DT_RANK + D_STATE + nS];
    }
    ys[row * D_INNER + d] = y;
  }
}

// -------------------------- gating: y=(ys+xc*D)*silu(z) --------------------

__global__ void k_gate(const float* __restrict__ ys,
                       const float* __restrict__ xc,
                       const float* __restrict__ xz,
                       const float* __restrict__ Dp,
                       u16* __restrict__ yb, int n) {
  int i = blockIdx.x * blockDim.x + threadIdx.x;
  if (i >= n) return;
  int row = i / D_INNER, d = i - row * D_INNER;
  float z = xz[(size_t)row * (2 * D_INNER) + D_INNER + d];
  float v = (ys[i] + xc[i] * Dp[d]) * (z / (1.f + __expf(-z)));
  yb[i] = f2bf(v);
}

// ------------------------------ mean pool ----------------------------------

__global__ void k_pool(const float* __restrict__ seq, float* __restrict__ pooled) {
  int i = blockIdx.x * blockDim.x + threadIdx.x;
  if (i >= BATCH * D_MODEL) return;
  int b = i / D_MODEL, e = i - b * D_MODEL;
  float s = 0.f;
  for (int t = 0; t < SEQ; ++t)
    s += seq[((size_t)b * SEQ + t) * D_MODEL + e];
  pooled[i] = s * (1.f / SEQ);
}

// ---------------------------- projection head ------------------------------

__global__ void k_head(const float* __restrict__ pooled,
                       const float* __restrict__ pw,
                       const float* __restrict__ pb,
                       float* __restrict__ out) {
  int i = blockIdx.x * blockDim.x + threadIdx.x;
  if (i >= BATCH * EMBED_DIM) return;
  int b = i / EMBED_DIM, j = i - b * EMBED_DIM;
  float s = pb[j];
  const float* pr = pooled + (size_t)b * D_MODEL;
  const float* wr = pw + (size_t)j * D_MODEL;
#pragma unroll 8
  for (int e = 0; e < D_MODEL; ++e) s += pr[e] * wr[e];
  out[i] = s;
}

// ------------------------------- launcher ----------------------------------

extern "C" void kernel_launch(void* const* d_in, const int* in_sizes, int n_in,
                              void* d_out, int out_size, void* d_ws, size_t ws_size,
                              hipStream_t stream) {
  (void)in_sizes; (void)n_in; (void)out_size; (void)ws_size;
  const int*   ids  = (const int*)  d_in[0];
  const float* emb  = (const float*)d_in[1];
  const float* inw  = (const float*)d_in[2];
  const float* cw   = (const float*)d_in[3];
  const float* cb   = (const float*)d_in[4];
  const float* xpw  = (const float*)d_in[5];
  const float* dtw  = (const float*)d_in[6];
  const float* dtb  = (const float*)d_in[7];
  const float* Alog = (const float*)d_in[8];
  const float* Dp   = (const float*)d_in[9];
  const float* outw = (const float*)d_in[10];
  const float* pw   = (const float*)d_in[11];
  const float* pb   = (const float*)d_in[12];
  float* out = (float*)d_out;

  // workspace carve-out (256B aligned slices)
  char* p = (char*)d_ws;
  auto alloc = [&](size_t bytes) -> char* {
    char* r = p; p += (bytes + 255) & ~(size_t)255; return r;
  };
  u16*   x_bf    = (u16*)  alloc((size_t)ROWS * D_MODEL * 2);
  u16*   inw_bf  = (u16*)  alloc((size_t)2 * D_INNER * D_MODEL * 2);
  u16*   xpw_bf  = (u16*)  alloc((size_t)XDBL_W * D_INNER * 2);
  u16*   outw_bf = (u16*)  alloc((size_t)D_MODEL * D_INNER * 2);
  float* xz      = (float*)alloc((size_t)ROWS * 2 * D_INNER * 4);
  float* xc      = (float*)alloc((size_t)ROWS * D_INNER * 4);
  u16*   xc_bf   = (u16*)  alloc((size_t)ROWS * D_INNER * 2);
  float* xdbl    = (float*)alloc((size_t)ROWS * XDBL_W * 4);
  float* delta   = (float*)alloc((size_t)ROWS * D_INNER * 4);
  float* ys      = (float*)alloc((size_t)ROWS * D_INNER * 4);
  u16*   y_bf    = (u16*)  alloc((size_t)ROWS * D_INNER * 2);
  float* seq     = (float*)alloc((size_t)ROWS * D_MODEL * 4);
  float* pooled  = (float*)alloc((size_t)BATCH * D_MODEL * 4);

  const int TB = 256;
  auto nb = [](int n) { return (n + 255) / 256; };

  // 1) weight casts to bf16
  {
    int n1 = 2 * D_INNER * D_MODEL;
    k_f32_to_bf16<<<nb(n1), TB, 0, stream>>>(inw, inw_bf, n1);
    int n2 = XDBL_W * D_INNER;
    k_f32_to_bf16<<<nb(n2), TB, 0, stream>>>(xpw, xpw_bf, n2);
    int n3 = D_MODEL * D_INNER;
    k_f32_to_bf16<<<nb(n3), TB, 0, stream>>>(outw, outw_bf, n3);
  }

  // 2) embedding gather -> bf16 activations
  {
    int n = ROWS * D_MODEL;
    k_gather_embed<<<nb(n), TB, 0, stream>>>(ids, emb, x_bf, n);
  }

  // 3) in_proj: [8192,768] x [3072,768]^T -> xz [8192,3072]   (TDM path)
  k_gemm_bf16_wmma_tdm<<<dim3(ROWS / 256, (2 * D_INNER) / 64), TB, 0, stream>>>(
      x_bf, inw_bf, xz, ROWS, 2 * D_INNER, D_MODEL);

  // 4) depthwise causal conv + SiLU
  {
    int n = ROWS * D_INNER;
    k_conv_silu<<<nb(n), TB, 0, stream>>>(xz, cw, cb, xc, xc_bf, n);
  }

  // 5) x_proj: [8192,1536] x [80,1536]^T -> x_dbl [8192,80]   (direct path)
  k_gemm_bf16_wmma<<<dim3(ROWS / 128, (XDBL_W + 63) / 64), TB, 0, stream>>>(
      xc_bf, xpw_bf, xdbl, ROWS, XDBL_W, D_INNER);

  // 6) delta = softplus(dt @ dt_proj_w^T + b)
  {
    int n = ROWS * D_INNER;
    k_delta<<<nb(n), TB, 0, stream>>>(xdbl, dtw, dtb, delta, n);
  }

  // 7) selective scan (sequential over L, parallel over B*D_INNER)
  k_scan<<<(BATCH * D_INNER + TB - 1) / TB, TB, 0, stream>>>(
      delta, xdbl, xc, Alog, ys);

  // 8) gate: y = (ys + xc*D) * silu(z)  -> bf16
  {
    int n = ROWS * D_INNER;
    k_gate<<<nb(n), TB, 0, stream>>>(ys, xc, xz, Dp, y_bf, n);
  }

  // 9) out_proj: [8192,1536] x [768,1536]^T -> seq [8192,768]  (TDM path)
  k_gemm_bf16_wmma_tdm<<<dim3(ROWS / 256, D_MODEL / 64), TB, 0, stream>>>(
      y_bf, outw_bf, seq, ROWS, D_MODEL, D_INNER);

  // 10) mean pool over L
  k_pool<<<(BATCH * D_MODEL + TB - 1) / TB, TB, 0, stream>>>(seq, pooled);

  // 11) projection head -> out [4,256] fp32
  k_head<<<(BATCH * EMBED_DIM + TB - 1) / TB, TB, 0, stream>>>(pooled, pw, pb, out);
}
